// BitNetBlock_51230369907328
// MI455X (gfx1250) — 16-bit and 8-bit matrix operand order repaired and checked by reading, compiled, not run on hardware
//
#include <hip/hip_runtime.h>
#include <stdint.h>


typedef int          v8i   __attribute__((ext_vector_type(8)));
typedef float        v8f   __attribute__((ext_vector_type(8)));
typedef _Float16     v16h  __attribute__((ext_vector_type(16)));
typedef unsigned int u32x4 __attribute__((ext_vector_type(4)));
typedef int          i32x4 __attribute__((ext_vector_type(4)));
typedef int          i32x8 __attribute__((ext_vector_type(8)));

union I8x8 { v8i  v; int      i[8];  int2 d2[4]; int4 d4[2]; };
union F8x  { v8f  v; float    f[8]; };
union H16x { v16h v; _Float16 h[16]; int4 d4[2]; };

static constexpr int   BB   = 2;
static constexpr int   NN   = 2048;
static constexpr int   CC   = 1024;
static constexpr int   HH   = 16;
static constexpr int   DD   = 64;
static constexpr int   HID  = 4096;
static constexpr int   MM   = BB * NN;
static constexpr float EPSF = 1e-5f;

__device__ inline void tdm_load_2d_i8(uint32_t lds_off, const void* gaddr,
                                      uint32_t dim0, uint32_t dim1,
                                      uint32_t tile0, uint32_t tile1,
                                      uint64_t stride0) {
  const uint64_t ga = (uint64_t)gaddr;
  u32x4 g0;
  g0[0] = 1u;
  g0[1] = lds_off;
  g0[2] = (uint32_t)ga;
  g0[3] = (uint32_t)((ga >> 32) & 0x1FFFFFFu) | (2u << 30);
  i32x8 g1;
  g1[0] = 0;
  g1[1] = (int)((dim0 & 0xFFFFu) << 16);
  g1[2] = (int)((dim0 >> 16) | ((dim1 & 0xFFFFu) << 16));
  g1[3] = (int)((dim1 >> 16) | (tile0 << 16));
  g1[4] = (int)(tile1 & 0xFFFFu);
  g1[5] = (int)(uint32_t)stride0;
  g1[6] = (int)(uint32_t)(stride0 >> 32);
  g1[7] = 0;
  const i32x4 z4 = {0, 0, 0, 0};
  const i32x8 z8 = {0, 0, 0, 0, 0, 0, 0, 0};
  __builtin_amdgcn_tensor_load_to_lds(g0, g1, z4, z4, z8, 0);
}

__global__ void bn_zero4(float* wsum) {
  if (threadIdx.x < 4) wsum[threadIdx.x] = 0.f;
}

__global__ __launch_bounds__(256) void bn_absmean(const float* __restrict__ w,
                                                  unsigned long long n,
                                                  float* __restrict__ slot) {
  __shared__ float red[256];
  float s = 0.f;
  for (unsigned long long i = (unsigned long long)blockIdx.x * 256 + threadIdx.x;
       i < n; i += (unsigned long long)gridDim.x * 256)
    s += fabsf(w[i]);
  red[threadIdx.x] = s; __syncthreads();
  for (int o = 128; o > 0; o >>= 1) {
    if ((int)threadIdx.x < o) red[threadIdx.x] += red[threadIdx.x + o];
    __syncthreads();
  }
  if (threadIdx.x == 0) atomicAdd(slot, red[0]);
}

__global__ void bn_finalize_ws(float* __restrict__ wscale,
                               const float* __restrict__ wsum,
                               float c0, float c1, float c2, float c3) {
  float c[4] = {c0, c1, c2, c3};
  int i = threadIdx.x;
  if (i < 4) wscale[i] = 1.f / fmaxf(wsum[i] / c[i], EPSF);
}

__global__ __launch_bounds__(256) void bn_quant_w(const float* __restrict__ w,
                                                  int8_t* __restrict__ wq,
                                                  const float* __restrict__ wscale,
                                                  unsigned long long n) {
  float s = *wscale;
  for (unsigned long long i = (unsigned long long)blockIdx.x * 256 + threadIdx.x;
       i < n; i += (unsigned long long)gridDim.x * 256) {
    float q = rintf(w[i] * s);
    q = fminf(1.f, fmaxf(-1.f, q));
    wq[i] = (int8_t)q;
  }
}

__global__ __launch_bounds__(256) void bn_ln_quant(const float* __restrict__ x,
                                                   const float* __restrict__ g,
                                                   const float* __restrict__ be,
                                                   int8_t* __restrict__ xq,
                                                   float* __restrict__ ascale) {
  __shared__ float red[256];
  const int r = blockIdx.x, t = threadIdx.x;
  const float* row = x + (size_t)r * CC;
  float v0 = row[t], v1 = row[t + 256], v2 = row[t + 512], v3 = row[t + 768];
  red[t] = v0 + v1 + v2 + v3; __syncthreads();
  for (int o = 128; o > 0; o >>= 1) { if (t < o) red[t] += red[t + o]; __syncthreads(); }
  float mu = red[0] * (1.f / CC); __syncthreads();
  float d0 = v0 - mu, d1 = v1 - mu, d2 = v2 - mu, d3 = v3 - mu;
  red[t] = d0 * d0 + d1 * d1 + d2 * d2 + d3 * d3; __syncthreads();
  for (int o = 128; o > 0; o >>= 1) { if (t < o) red[t] += red[t + o]; __syncthreads(); }
  float rstd = rsqrtf(red[0] * (1.f / CC) + EPSF); __syncthreads();
  float y0 = d0 * rstd * g[t]       + be[t];
  float y1 = d1 * rstd * g[t + 256] + be[t + 256];
  float y2 = d2 * rstd * g[t + 512] + be[t + 512];
  float y3 = d3 * rstd * g[t + 768] + be[t + 768];
  red[t] = fmaxf(fmaxf(fabsf(y0), fabsf(y1)), fmaxf(fabsf(y2), fabsf(y3)));
  __syncthreads();
  for (int o = 128; o > 0; o >>= 1) { if (t < o) red[t] = fmaxf(red[t], red[t + o]); __syncthreads(); }
  float sc = 127.f / fmaxf(red[0], EPSF);
  int8_t* orow = xq + (size_t)r * CC;
  orow[t]       = (int8_t)fminf(127.f, fmaxf(-128.f, rintf(y0 * sc)));
  orow[t + 256] = (int8_t)fminf(127.f, fmaxf(-128.f, rintf(y1 * sc)));
  orow[t + 512] = (int8_t)fminf(127.f, fmaxf(-128.f, rintf(y2 * sc)));
  orow[t + 768] = (int8_t)fminf(127.f, fmaxf(-128.f, rintf(y3 * sc)));
  if (t == 0) ascale[r] = sc;
}

__global__ __launch_bounds__(256) void bn_row_quant(const float* __restrict__ src,
                                                    int8_t* __restrict__ dst,
                                                    float* __restrict__ ascale, int L) {
  __shared__ float red[256];
  const int r = blockIdx.x, t = threadIdx.x;
  const float* row = src + (size_t)r * L;
  float amax = 0.f;
  for (int i = t; i < L; i += 256) amax = fmaxf(amax, fabsf(row[i]));
  red[t] = amax; __syncthreads();
  for (int o = 128; o > 0; o >>= 1) { if (t < o) red[t] = fmaxf(red[t], red[t + o]); __syncthreads(); }
  float sc = 127.f / fmaxf(red[0], EPSF);
  int8_t* orow = dst + (size_t)r * L;
  for (int i = t; i < L; i += 256)
    orow[i] = (int8_t)fminf(127.f, fmaxf(-128.f, rintf(row[i] * sc)));
  if (t == 0) ascale[r] = sc;
}

__device__ inline v8i fetch_a8(const int8_t* p, int half) {
  I8x8 u;
  u.d4[0] = *(const int4*)(p +  0 + half * 16);
  u.d4[1] = *(const int4*)(p + 32 + half * 16);
  return u.v;
}
__device__ inline v8i fetch_b8(const int8_t* p, int half) {
  I8x8 u;
  u.d4[0] = *(const int4*)(p +  0 + half * 16);
  u.d4[1] = *(const int4*)(p + 32 + half * 16);
  return u.v;
}

template <int MODE>
__global__ __launch_bounds__(256) void bn_gemm_i8(
    const int8_t* __restrict__ Aq, const float* __restrict__ ascale,
    const int8_t* __restrict__ Wq, const float* __restrict__ wscale,
    const float* __restrict__ bias, int Nout, int K,
    float* __restrict__ out, const float* __restrict__ resid,
    _Float16* __restrict__ qf, _Float16* __restrict__ kf,
    _Float16* __restrict__ vT) {
  __shared__ __align__(16) int8_t sA[2][64 * 64];
  __shared__ __align__(16) int8_t sB[2][128 * 64];

  const int lane = threadIdx.x & 31, wave = threadIdx.x >> 5;
  const int half = lane >> 4, lr = lane & 15;
  const int wm = wave >> 2, wn = wave & 3;
  const int nb = Nout >> 7;
  const int mt = blockIdx.x / nb, nt = blockIdx.x % nb;
  const int mBase = mt * 64, nBase = nt * 128;
  const int m0 = mBase + wm * 32, n0 = nBase + wn * 32;

  const uint32_t offA[2] = { (uint32_t)(uintptr_t)&sA[0][0],
                             (uint32_t)(uintptr_t)&sA[1][0] };
  const uint32_t offB[2] = { (uint32_t)(uintptr_t)&sB[0][0],
                             (uint32_t)(uintptr_t)&sB[1][0] };

  I8x8 acc[2][2];
  const v8i z8 = {0, 0, 0, 0, 0, 0, 0, 0};
#pragma unroll
  for (int i = 0; i < 2; ++i)
#pragma unroll
    for (int j = 0; j < 2; ++j) acc[i][j].v = z8;

  if (wave == 0) {
    tdm_load_2d_i8(offA[0], Aq + (size_t)mBase * K, K, MM,   64,  64, (uint64_t)K);
    tdm_load_2d_i8(offB[0], Wq + (size_t)nBase * K, K, Nout, 64, 128, (uint64_t)K);
    __builtin_amdgcn_s_wait_tensorcnt(0);
  }
  __syncthreads();

  int cur = 0;
  for (int k0 = 0; k0 < K; k0 += 64) {
    const bool pref = (k0 + 64) < K;
    if (wave == 0 && pref) {
      tdm_load_2d_i8(offA[cur ^ 1], Aq + (size_t)mBase * K + k0 + 64, K, MM,   64,  64, (uint64_t)K);
      tdm_load_2d_i8(offB[cur ^ 1], Wq + (size_t)nBase * K + k0 + 64, K, Nout, 64, 128, (uint64_t)K);
    }
    const int8_t* aT0 = &sA[cur][(wm * 32 + lr) * 64];
    const int8_t* aT1 = aT0 + 16 * 64;
    const int8_t* bT0 = &sB[cur][(wn * 32 + lr) * 64];
    const int8_t* bT1 = bT0 + 16 * 64;
    v8i ta0 = fetch_a8(aT0, half);
    v8i ta1 = fetch_a8(aT1, half);
    v8i tb0 = fetch_b8(bT0, half);
    v8i tb1 = fetch_b8(bT1, half);
    acc[0][0].v = __builtin_amdgcn_wmma_i32_16x16x64_iu8(true, ta0, true, tb0, acc[0][0].v, false, false);
    acc[0][1].v = __builtin_amdgcn_wmma_i32_16x16x64_iu8(true, ta0, true, tb1, acc[0][1].v, false, false);
    acc[1][0].v = __builtin_amdgcn_wmma_i32_16x16x64_iu8(true, ta1, true, tb0, acc[1][0].v, false, false);
    acc[1][1].v = __builtin_amdgcn_wmma_i32_16x16x64_iu8(true, ta1, true, tb1, acc[1][1].v, false, false);
    if (wave == 0 && pref) __builtin_amdgcn_s_wait_tensorcnt(0);
    __syncthreads();
    cur ^= 1;
  }

  const float ws = *wscale;
#pragma unroll
  for (int ti = 0; ti < 2; ++ti) {
#pragma unroll
    for (int v = 0; v < 8; ++v) {
      const int row = m0 + ti * 16 + v + half * 8;
      const float inv = 1.f / (ascale[row] * ws);
#pragma unroll
      for (int tj = 0; tj < 2; ++tj) {
        const int col = n0 + tj * 16 + lr;
        float val = (float)acc[ti][tj].i[v] * inv + bias[col];
        if constexpr (MODE == 0) {
          const int part = col >> 10, rem = col & 1023;
          const int head = rem >> 6, dd = rem & 63;
          const int bb = row >> 11, tok = row & 2047;
          if (part == 0)
            qf[((size_t)(bb * HH + head) * NN + tok) * DD + dd] = (_Float16)(val * 0.125f);
          else if (part == 1)
            kf[((size_t)(bb * HH + head) * NN + tok) * DD + dd] = (_Float16)val;
          else
            vT[((size_t)(bb * HH + head) * DD + dd) * NN + tok] = (_Float16)val;
        } else if constexpr (MODE == 1) {
          out[(size_t)row * Nout + col] = resid[(size_t)row * Nout + col] + val;
        } else {
          out[(size_t)row * Nout + col] = 0.5f * val * (1.f + erff(val * 0.70710678118654752f));
        }
      }
    }
  }
}

__global__ __launch_bounds__(256) void bn_attn(const _Float16* __restrict__ qf,
                                               const _Float16* __restrict__ kf,
                                               const _Float16* __restrict__ vT,
                                               float* __restrict__ o) {
  __shared__ __align__(16) _Float16 pbuf[8][16 * 32];
  const int lane = threadIdx.x & 31, wave = threadIdx.x >> 5;
  const int half = lane >> 4, lr = lane & 15;
  const int gid = blockIdx.x * 8 + wave;
  const int qt = gid & 127;
  const int bh = gid >> 7;
  const _Float16* qb = qf + (size_t)bh * NN * DD;
  const _Float16* kb = kf + (size_t)bh * NN * DD;
  const _Float16* vb = vT + (size_t)bh * DD * NN;

  H16x qa[2];
  const _Float16* qrow = qb + (size_t)(qt * 16 + lr) * DD;
#pragma unroll
  for (int t = 0; t < 2; ++t) {
    qa[t].d4[0] = *(const int4*)(qrow + t * 32 + half * 16);
    qa[t].d4[1] = *(const int4*)(qrow + t * 32 + half * 16 + 8);
  }

  H16x onesb;
#pragma unroll
  for (int i = 0; i < 16; ++i) onesb.h[i] = (_Float16)1.f;

  F8x oacc[4], lacc;
  float mrun = -3.0e38f;
#pragma unroll
  for (int dt = 0; dt < 4; ++dt)
#pragma unroll
    for (int v = 0; v < 8; ++v) oacc[dt].f[v] = 0.f;
#pragma unroll
  for (int v = 0; v < 8; ++v) lacc.f[v] = 0.f;

  const v8f zf = {0.f, 0.f, 0.f, 0.f, 0.f, 0.f, 0.f, 0.f};

  for (int k0 = 0; k0 < NN; k0 += 32) {
    F8x s[2];
#pragma unroll
    for (int nt2 = 0; nt2 < 2; ++nt2) {
      H16x bk0, bk1;
      const _Float16* krow = kb + (size_t)(k0 + nt2 * 16 + lr) * DD;
      bk0.d4[0] = *(const int4*)(krow + half * 16);
      bk0.d4[1] = *(const int4*)(krow + half * 16 + 8);
      bk1.d4[0] = *(const int4*)(krow + 32 + half * 16);
      bk1.d4[1] = *(const int4*)(krow + 32 + half * 16 + 8);
      s[nt2].v = __builtin_amdgcn_wmma_f32_16x16x32_f16(false, qa[0].v, false, bk0.v, (short)0, zf, false, false);
      s[nt2].v = __builtin_amdgcn_wmma_f32_16x16x32_f16(false, qa[1].v, false, bk1.v, (short)0, s[nt2].v, false, false);
    }
#pragma unroll
    for (int v = 0; v < 8; ++v) {
      const int prow = v + half * 8;
      pbuf[wave][prow * 32 + lr]      = (_Float16)s[0].f[v];
      pbuf[wave][prow * 32 + 16 + lr] = (_Float16)s[1].f[v];
    }
    H16x sl;
    const _Float16* prp = &pbuf[wave][lr * 32];
    sl.d4[0] = *(const int4*)(prp + half * 16);
    sl.d4[1] = *(const int4*)(prp + half * 16 + 8);
    float tmax = -3.0e38f;
#pragma unroll
    for (int i = 0; i < 16; ++i) tmax = fmaxf(tmax, (float)sl.h[i]);
    tmax = fmaxf(tmax, __shfl_xor(tmax, 16, 32));
    const float mnew = fmaxf(mrun, tmax);
    const float corr = __expf(mrun - mnew);
    mrun = mnew;
    H16x pa;
#pragma unroll
    for (int i = 0; i < 16; ++i) pa.h[i] = (_Float16)__expf((float)sl.h[i] - mnew);
    float corr_acc[8];
#pragma unroll
    for (int v = 0; v < 8; ++v) corr_acc[v] = __shfl(corr, v + half * 8, 32);
#pragma unroll
    for (int v = 0; v < 8; ++v) {
      lacc.f[v] *= corr_acc[v];
#pragma unroll
      for (int dt = 0; dt < 4; ++dt) oacc[dt].f[v] *= corr_acc[v];
    }
    lacc.v = __builtin_amdgcn_wmma_f32_16x16x32_f16(false, pa.v, false, onesb.v, (short)0, lacc.v, false, false);
#pragma unroll
    for (int dt = 0; dt < 4; ++dt) {
      H16x bv;
      const _Float16* vrow = vb + (size_t)(dt * 16 + lr) * NN + k0 + half * 16;
      bv.d4[0] = *(const int4*)(vrow);
      bv.d4[1] = *(const int4*)(vrow + 8);
      oacc[dt].v = __builtin_amdgcn_wmma_f32_16x16x32_f16(false, pa.v, false, bv.v, (short)0, oacc[dt].v, false, false);
    }
  }

  const int bb = bh >> 4, hh = bh & 15;
#pragma unroll
  for (int v = 0; v < 8; ++v) {
    const int tok = qt * 16 + v + half * 8;
    const size_t rowbase = ((size_t)(bb * NN + tok)) * CC + hh * DD;
    const float invl = 1.f / lacc.f[v];
#pragma unroll
    for (int dt = 0; dt < 4; ++dt)
      o[rowbase + dt * 16 + lr] = oacc[dt].f[v] * invl;
  }
}

extern "C" void kernel_launch(void* const* d_in, const int* in_sizes, int n_in,
                              void* d_out, int out_size, void* d_ws, size_t ws_size,
                              hipStream_t stream) {
  (void)in_sizes; (void)n_in; (void)out_size; (void)ws_size;
  const float* x      = (const float*)d_in[0];
  const float* g1     = (const float*)d_in[1];
  const float* be1    = (const float*)d_in[2];
  const float* g2     = (const float*)d_in[3];
  const float* be2    = (const float*)d_in[4];
  const float* w_qkv  = (const float*)d_in[5];
  const float* b_qkv  = (const float*)d_in[6];
  const float* w_proj = (const float*)d_in[7];
  const float* b_proj = (const float*)d_in[8];
  const float* w_fc1  = (const float*)d_in[9];
  const float* b_fc1  = (const float*)d_in[10];
  const float* w_fc2  = (const float*)d_in[11];
  const float* b_fc2  = (const float*)d_in[12];
  float* out = (float*)d_out;

  char* base = (char*)d_ws;
  size_t off = 0;
  auto alloc = [&](size_t sz) { size_t o = off; off = (off + sz + 255) & ~(size_t)255; return o; };
  const size_t n_qkv = (size_t)3 * CC * CC, n_proj = (size_t)CC * CC;
  const size_t n_fc1 = (size_t)HID * CC,  n_fc2  = (size_t)CC * HID;

  float*  wsum    = (float*)(base + alloc(16));
  float*  wscale4 = (float*)(base + alloc(16));
  int8_t* wq_qkv  = (int8_t*)(base + alloc(n_qkv));
  int8_t* wq_proj = (int8_t*)(base + alloc(n_proj));
  int8_t* wq_fc1  = (int8_t*)(base + alloc(n_fc1));
  int8_t* wq_fc2  = (int8_t*)(base + alloc(n_fc2));
  int8_t* xq      = (int8_t*)(base + alloc((size_t)MM * HID));
  float*  ascale  = (float*)(base + alloc((size_t)MM * 4));
  _Float16* qf16  = (_Float16*)(base + alloc((size_t)BB * HH * NN * DD * 2));
  _Float16* kf16  = (_Float16*)(base + alloc((size_t)BB * HH * NN * DD * 2));
  _Float16* vT16  = (_Float16*)(base + alloc((size_t)BB * HH * DD * NN * 2));
  float*  obuf    = (float*)(base + alloc((size_t)MM * CC * 4));
  float*  x1buf   = (float*)(base + alloc((size_t)MM * CC * 4));
  float*  hbuf    = (float*)(base + alloc((size_t)MM * HID * 4));

  bn_zero4<<<1, 64, 0, stream>>>(wsum);
  bn_absmean<<<256, 256, 0, stream>>>(w_qkv,  n_qkv,  wsum + 0);
  bn_absmean<<<256, 256, 0, stream>>>(w_proj, n_proj, wsum + 1);
  bn_absmean<<<256, 256, 0, stream>>>(w_fc1,  n_fc1,  wsum + 2);
  bn_absmean<<<256, 256, 0, stream>>>(w_fc2,  n_fc2,  wsum + 3);
  bn_finalize_ws<<<1, 64, 0, stream>>>(wscale4, wsum,
                                       (float)n_qkv, (float)n_proj,
                                       (float)n_fc1, (float)n_fc2);
  bn_quant_w<<<2048, 256, 0, stream>>>(w_qkv,  wq_qkv,  wscale4 + 0, n_qkv);
  bn_quant_w<<<2048, 256, 0, stream>>>(w_proj, wq_proj, wscale4 + 1, n_proj);
  bn_quant_w<<<2048, 256, 0, stream>>>(w_fc1,  wq_fc1,  wscale4 + 2, n_fc1);
  bn_quant_w<<<2048, 256, 0, stream>>>(w_fc2,  wq_fc2,  wscale4 + 3, n_fc2);

  bn_ln_quant<<<MM, 256, 0, stream>>>(x, g1, be1, xq, ascale);
  bn_gemm_i8<0><<<(MM / 64) * (3 * CC / 128), 256, 0, stream>>>(
      xq, ascale, wq_qkv, wscale4 + 0, b_qkv, 3 * CC, CC,
      nullptr, nullptr, qf16, kf16, vT16);

  bn_attn<<<(BB * HH * (NN / 16)) / 8, 256, 0, stream>>>(qf16, kf16, vT16, obuf);

  bn_row_quant<<<MM, 256, 0, stream>>>(obuf, xq, ascale, CC);
  bn_gemm_i8<1><<<(MM / 64) * (CC / 128), 256, 0, stream>>>(
      xq, ascale, wq_proj, wscale4 + 1, b_proj, CC, CC,
      x1buf, x, nullptr, nullptr, nullptr);

  bn_ln_quant<<<MM, 256, 0, stream>>>(x1buf, g2, be2, xq, ascale);
  bn_gemm_i8<2><<<(MM / 64) * (HID / 128), 256, 0, stream>>>(
      xq, ascale, wq_fc1, wscale4 + 2, b_fc1, HID, CC,
      hbuf, nullptr, nullptr, nullptr, nullptr);

  bn_row_quant<<<MM, 256, 0, stream>>>(hbuf, xq, ascale, HID);
  bn_gemm_i8<1><<<(MM / 64) * (CC / 128), 256, 0, stream>>>(
      xq, ascale, wq_fc2, wscale4 + 3, b_fc2, CC, HID,
      out, x1buf, nullptr, nullptr, nullptr);
}
